// Coattention_70342974373949
// MI455X (gfx1250) — matrix operand order repaired and checked by reading, not compiled, not run on hardware
//
#include <hip/hip_runtime.h>

typedef unsigned int u32;
typedef unsigned short u16;
typedef __attribute__((ext_vector_type(16))) __bf16 v16bf;
typedef __attribute__((ext_vector_type(8)))  float  v8f;
typedef __attribute__((ext_vector_type(4)))  int    v4i_t;

#define D_DIM 768
#define BATCH 32
#define TM 128
#define TN 128
#define TK 32
#define LDSS 40   // LDS row stride (halves): conflict-free frag reads, 16B-aligned rows

#if defined(__gfx1250__) && __has_builtin(__builtin_amdgcn_global_load_async_to_lds_b128)
#define HAVE_ASYNC_LDS 1
#else
#define HAVE_ASYNC_LDS 0
#endif

__device__ __forceinline__ u16 f2bf(float f) {
  u32 u = __builtin_bit_cast(u32, f);
  u += 0x7FFFu + ((u >> 16) & 1u);
  return (u16)(u >> 16);
}
__device__ __forceinline__ float bf2f(u16 h) {
  u32 u = ((u32)h) << 16;
  return __builtin_bit_cast(float, u);
}

union FragBF {
  uint4 q[2];
  v16bf v;
};

// Copy 16 bytes global -> LDS. Async (ASYNCcnt-tracked) when available.
__device__ __forceinline__ void stage16(const u16* __restrict__ g, u16* l) {
#if HAVE_ASYNC_LDS
  __attribute__((address_space(1))) v4i_t* gp =
      (__attribute__((address_space(1))) v4i_t*)
      (__attribute__((address_space(1))) void*)(void*)g;
  __attribute__((address_space(3))) v4i_t* lp =
      (__attribute__((address_space(3))) v4i_t*)
      (__attribute__((address_space(3))) void*)l;
  __builtin_amdgcn_global_load_async_to_lds_b128(gp, lp, 0, 0);
#else
  *(uint4*)l = *(const uint4*)g;
#endif
}

__device__ __forceinline__ void wait_async_lds() {
#if HAVE_ASYNC_LDS
#if __has_builtin(__builtin_amdgcn_s_wait_asynccnt)
  __builtin_amdgcn_s_wait_asynccnt(0);
#else
  asm volatile("s_wait_asynccnt 0x0" ::: "memory");
#endif
#endif
}

// ---------------------------------------------------------------------------
// Batched NT GEMM: C[b] = A[b] (MxK, bf16 row-major) @ Bt^T  (Bt: NxK row-major)
//                  (+ bias per column) (+ bf16 residual) -> bf16 or fp32 out
// Bt is per-batch if bBatched, else shared (weights, used directly: x@W^T).
// Both operand tiles are pure row-copies: async global->LDS, double-buffered,
// one barrier per K-step.
// ---------------------------------------------------------------------------
__global__ __launch_bounds__(256) void gemm_nt_bf16_wmma(
    const u16* __restrict__ A, const u16* __restrict__ Bt,
    u16* __restrict__ Cb, float* __restrict__ Cf,
    const float* __restrict__ bias, const u16* __restrict__ resid,
    int bBatched)
{
  __shared__ __align__(16) u16 As[2][TM * LDSS];
  __shared__ __align__(16) u16 Bs[2][TN * LDSS];

  const int M = D_DIM, N = D_DIM, K = D_DIM;
  const int b = blockIdx.y;
  const int ntiles = N / TN;
  const int tm = (blockIdx.x / ntiles) * TM;
  const int tn = (blockIdx.x % ntiles) * TN;

  const u16* Ab  = A  + (size_t)b * M * K;
  const u16* Btb = Bt + (bBatched ? (size_t)b * N * K : 0);

  const int t = threadIdx.x;
  const int lane = t & 31;
  const int wave = t >> 5;
  const int waveM = wave & 3;   // 4 row groups of 32
  const int waveN = wave >> 2;  // 2 col groups of 64
  const int laneLo = lane & 15;
  const int laneHi = lane >> 4; // 0/1

  v8f acc[2][4];
#pragma unroll
  for (int i = 0; i < 2; ++i)
#pragma unroll
    for (int j = 0; j < 4; ++j)
#pragma unroll
      for (int r = 0; r < 8; ++r) acc[i][j][r] = 0.0f;

  // staging: 128 rows x two 16-half segments, identical for A and Bt tiles
  const int srow = t >> 1, sseg = t & 1;

  auto stageTiles = [&](int buf, int k0) {
    const u16* ga = Ab + (size_t)(tm + srow) * K + k0 + sseg * 16;
    u16* la = &As[buf][srow * LDSS + sseg * 16];
    stage16(ga, la);
    stage16(ga + 8, la + 8);
    const u16* gb = Btb + (size_t)(tn + srow) * K + k0 + sseg * 16;
    u16* lb = &Bs[buf][srow * LDSS + sseg * 16];
    stage16(gb, lb);
    stage16(gb + 8, lb + 8);
  };

  stageTiles(0, 0);

  for (int k0 = 0; k0 < K; k0 += TK) {
    const int cur = (k0 >> 5) & 1;
    wait_async_lds();      // our wave's copies into buf[cur] are done
    __syncthreads();       // everyone's copies done; prior reads of buf[cur^1] done
    if (k0 + TK < K) stageTiles(cur ^ 1, k0 + TK);  // prefetch next tile

    // --- fragments per ISA 16-bit layouts ---
    FragBF af[2];
#pragma unroll
    for (int i = 0; i < 2; ++i) {
      const int mr = waveM * 32 + i * 16 + laneLo;
      const int kb = laneHi * 8;  // lanes>=16 start at K=8
      af[i].q[0] = *(const uint4*)&As[cur][mr * LDSS + kb];       // K kb..kb+7
      af[i].q[1] = *(const uint4*)&As[cur][mr * LDSS + kb + 16];  // K kb+16..kb+23
    }
    FragBF bfm[4];
#pragma unroll
    for (int j = 0; j < 4; ++j) {
      const int nn = waveN * 64 + j * 16 + laneLo;
      const int kb = laneHi * 8;  // lanes>=16 start at K=8
      bfm[j].q[0] = *(const uint4*)&Bs[cur][nn * LDSS + kb];
      bfm[j].q[1] = *(const uint4*)&Bs[cur][nn * LDSS + kb + 16];
    }

#pragma unroll
    for (int i = 0; i < 2; ++i)
#pragma unroll
      for (int j = 0; j < 4; ++j)
        acc[i][j] = __builtin_amdgcn_wmma_f32_16x16x32_bf16(
            false, af[i].v, false, bfm[j].v, (short)0, acc[i][j], false, false);
  }

  // --- epilogue: C layout VGPR r <-> M=r (lanes 0-15) / M=8+r (lanes 16-31) ---
  const size_t bOff = (size_t)b * M * N;
#pragma unroll
  for (int i = 0; i < 2; ++i) {
#pragma unroll
    for (int j = 0; j < 4; ++j) {
      const int gm = tm + waveM * 32 + i * 16 + laneHi * 8;
      const int gn = tn + waveN * 64 + j * 16 + laneLo;
      const float bv = bias ? bias[gn] : 0.0f;
#pragma unroll
      for (int r = 0; r < 8; ++r) {
        const int row = gm + r;
        float v = acc[i][j][r] + bv;
        if (resid) v += bf2f(resid[bOff + (size_t)row * N + gn]);
        if (Cf) Cf[bOff + (size_t)row * N + gn] = v;
        else    Cb[bOff + (size_t)row * N + gn] = f2bf(v);
      }
    }
  }
}

// ---------------------------------------------------------------------------
// Row softmax over last dim (N=768), in-place on bf16. One block per row.
// ---------------------------------------------------------------------------
__global__ __launch_bounds__(256) void softmax_rows_bf16(u16* __restrict__ data)
{
  const int N = D_DIM;
  __shared__ float red[8];
  u16* p = data + (size_t)blockIdx.x * N;
  const int t = threadIdx.x, lane = t & 31, wv = t >> 5;

  float v[3];
  float m = -3.4e38f;
#pragma unroll
  for (int i = 0; i < 3; ++i) { v[i] = bf2f(p[t + i * 256]); m = fmaxf(m, v[i]); }
#pragma unroll
  for (int off = 16; off > 0; off >>= 1) m = fmaxf(m, __shfl_xor(m, off, 32));
  if (lane == 0) red[wv] = m;
  __syncthreads();
  float mAll = red[0];
#pragma unroll
  for (int i = 1; i < 8; ++i) mAll = fmaxf(mAll, red[i]);
  __syncthreads();

  float s = 0.0f;
#pragma unroll
  for (int i = 0; i < 3; ++i) { v[i] = __expf(v[i] - mAll); s += v[i]; }
#pragma unroll
  for (int off = 16; off > 0; off >>= 1) s += __shfl_xor(s, off, 32);
  if (lane == 0) red[wv] = s;
  __syncthreads();
  float sAll = 0.0f;
#pragma unroll
  for (int i = 0; i < 8; ++i) sAll += red[i];
  const float inv = 1.0f / sAll;
#pragma unroll
  for (int i = 0; i < 3; ++i) p[t + i * 256] = f2bf(v[i] * inv);
}

// ---------------------------------------------------------------------------
// Batched bf16 transpose: out[b][j][i] = in[b][i][j], 64x64 LDS tiles.
// ---------------------------------------------------------------------------
__global__ __launch_bounds__(256) void transpose_bf16(const u16* __restrict__ in,
                                                      u16* __restrict__ out)
{
  __shared__ u16 tile[64][68];
  const int b = blockIdx.y;
  const int tX = blockIdx.x % (D_DIM / 64);
  const int tY = blockIdx.x / (D_DIM / 64);
  const u32* in32  = (const u32*)(in  + (size_t)b * D_DIM * D_DIM);
  u32*       out32 = (u32*)      (out + (size_t)b * D_DIM * D_DIM);
  const int cx = threadIdx.x & 31, ry = threadIdx.x >> 5;

#pragma unroll
  for (int i = 0; i < 8; ++i) {
    const int row = ry + i * 8;
    u32 w = in32[(size_t)(tY * 64 + row) * (D_DIM / 2) + tX * 32 + cx];
    tile[cx * 2][row]     = (u16)w;
    tile[cx * 2 + 1][row] = (u16)(w >> 16);
  }
  __syncthreads();
#pragma unroll
  for (int i = 0; i < 8; ++i) {
    const int col = ry + i * 8;
    u32 lo = tile[col][cx * 2];
    u32 hi = tile[col][cx * 2 + 1];
    out32[(size_t)(tX * 64 + col) * (D_DIM / 2) + tY * 32 + cx] = lo | (hi << 16);
  }
}

// fp32 -> bf16 elementwise convert
__global__ __launch_bounds__(256) void f32_to_bf16(const float* __restrict__ in,
                                                   u16* __restrict__ out, int n)
{
  int i = blockIdx.x * 256 + threadIdx.x;
  if (i < n) out[i] = f2bf(in[i]);
}

// ---------------------------------------------------------------------------
extern "C" void kernel_launch(void* const* d_in, const int* in_sizes, int n_in,
                              void* d_out, int out_size, void* d_ws, size_t ws_size,
                              hipStream_t stream)
{
  (void)in_sizes; (void)n_in; (void)out_size; (void)ws_size;
  const float* L  = (const float*)d_in[0];
  const float* I  = (const float*)d_in[1];
  const float* Wc = (const float*)d_in[2];
  const float* bc = (const float*)d_in[3];
  const float* Ws = (const float*)d_in[4];
  const float* bs = (const float*)d_in[5];
  const float* Wx = (const float*)d_in[6];
  const float* bx = (const float*)d_in[7];
  float* out = (float*)d_out;

  const size_t WSZ  = (size_t)D_DIM * D_DIM * sizeof(u16);          // one weight matrix
  const size_t SLOT = (size_t)BATCH * D_DIM * D_DIM * sizeof(u16);  // one activation buf

  char* ws  = (char*)d_ws;
  u16* wcb = (u16*)(ws);
  u16* wsb = (u16*)(ws + WSZ);
  u16* wxb = (u16*)(ws + 2 * WSZ);
  u16* s0  = (u16*)(ws + 3 * WSZ);
  u16* s1  = (u16*)(ws + 3 * WSZ + 1 * SLOT);
  u16* s2  = (u16*)(ws + 3 * WSZ + 2 * SLOT);
  u16* s3  = (u16*)(ws + 3 * WSZ + 3 * SLOT);
  u16* s4  = (u16*)(ws + 3 * WSZ + 4 * SLOT);

  const int NEL = BATCH * D_DIM * D_DIM;
  const int WEL = D_DIM * D_DIM;
  const dim3 gGemm(36, BATCH);                  // (768/128)^2 tiles x batch
  const dim3 gTr(144, BATCH);                   // (768/64)^2 tiles x batch
  const int  gSm = BATCH * D_DIM;               // one block per row

  auto nt = [&](const u16* A, const u16* Bt, int bBat, const float* bias,
                const u16* resid, u16* outB, float* outF) {
    gemm_nt_bf16_wmma<<<gGemm, 256, 0, stream>>>(A, Bt, outB, outF, bias, resid, bBat);
  };
  auto tr = [&](const u16* in, u16* o) {
    transpose_bf16<<<gTr, 256, 0, stream>>>(in, o);
  };
  auto sm = [&](u16* p) { softmax_rows_bf16<<<gSm, 256, 0, stream>>>(p); };

  // converts (weights used untransposed: x @ W^T == NT(x, W))
  f32_to_bf16<<<(WEL + 255) / 256, 256, 0, stream>>>(Wc, wcb, WEL);
  f32_to_bf16<<<(WEL + 255) / 256, 256, 0, stream>>>(Ws, wsb, WEL);
  f32_to_bf16<<<(WEL + 255) / 256, 256, 0, stream>>>(Wx, wxb, WEL);
  f32_to_bf16<<<(NEL + 255) / 256, 256, 0, stream>>>(L, s0, NEL);
  f32_to_bf16<<<(NEL + 255) / 256, 256, 0, stream>>>(I, s1, NEL);

  // ---- coattention block ----
  nt(s0, wcb, 0, bc, nullptr, s2, nullptr);   // lo  = L @ Wc^T + bc
  nt(s1, wcb, 0, bc, nullptr, s3, nullptr);   // io  = I @ Wc^T + bc     [live to end]
  tr(s2, s0);                                 // loT
  nt(s0, s3, 1, nullptr, nullptr, s1, nullptr); // S1 = loT @ io^T (= loT@ioT)
  sm(s1);                                     // A1
  tr(s1, s2);                                 // A1T
  nt(s2, s3, 1, nullptr, nullptr, s1, nullptr); // co1T = A1T @ io^T (= (io@A1)^T)
  nt(s1, wcb, 0, bc, s0, s2, nullptr);        // co = co1T @ Wc^T + bc + loT

  // ---- self-attention block ----
  nt(s2, wsb, 0, bs, nullptr, s0, nullptr);   // P  = co @ Ws^T + bs  (P = saT^T)
  tr(s0, s1);                                 // saT
  nt(s1, s0, 1, nullptr, nullptr, s4, nullptr); // S2 = saT @ P^T (= saT@saT)
  sm(s4);                                     // A2
  tr(s2, s0);                                 // coT
  nt(s0, s4, 1, nullptr, nullptr, s1, nullptr); // sa1T = coT @ A2^T (= (A2@co)^T)
  nt(s1, wsb, 0, bs, s2, s0, nullptr);        // sa = sa1T @ Ws^T + bs + co

  // ---- cross-attention block ----
  nt(s0, wxb, 0, bx, nullptr, s1, nullptr);   // Q  = sa @ Wx^T + bx  (Q = xaT^T)
  tr(s1, s2);                                 // xaT
  nt(s2, s3, 1, nullptr, nullptr, s1, nullptr); // S3 = xaT @ io^T (= xaT@ioT)
  sm(s1);                                     // A3
  nt(s3, s1, 1, nullptr, nullptr, s2, nullptr); // xa1T = io @ A3^T (= (A3@ioT)^T)
  nt(s2, wxb, 0, bx, s0, nullptr, out);       // out = xa1T @ Wx^T + bx + sa (fp32)
}
